// MoELayer_8950711846003
// MI455X (gfx1250) — matrix operand order repaired and checked by reading, not compiled, not run on hardware
//
#include <hip/hip_runtime.h>
#include <hip/hip_bf16.h>

#define DIM   1024
#define HID   4096
#define NE    8
#define NTOK  8192
#define NC    (NTOK * DIM)      // 8,388,608 combined-output floats
#define CAP   16384             // per-expert routing capacity (worst case)
#define TM    32                // tokens per FFN block (two 16-row WMMA tiles)
#define XP    1032              // xbuf pitch (ushort): 2064B, mult 16, bank step 4
#define HP    136               // hchunk pitch (ushort): 272B, mult 16, bank step 4
#define WP    36                // weight-stage pitch (ushort): 72B, 8B-aligned rows

typedef __bf16 v16bf __attribute__((ext_vector_type(16)));
typedef float  v8f   __attribute__((ext_vector_type(8)));
typedef unsigned long long ull;
typedef unsigned short ushortT;

union ABFrag {
    v16bf v;
    uint4 q[2];
    ull   d[4];
    ushortT s[16];
};

// fp32 -> bf16, round-half-up (2 VALU)
__device__ __forceinline__ ushortT f2bf(float f) {
    return (ushortT)((__float_as_uint(f) + 0x8000u) >> 16);
}

// pack two fp32 -> bf16x2 (low = a, high = b): 2 adds + 1 v_perm_b32
__device__ __forceinline__ unsigned pk2bf(float a, float b) {
    unsigned ua = __float_as_uint(a) + 0x8000u;
    unsigned ub = __float_as_uint(b) + 0x8000u;
    // bytes: dst0=a.b2, dst1=a.b3, dst2=b.b2, dst3=b.b3  (src1=a, src0=b)
    return __builtin_amdgcn_perm(ub, ua, 0x07060302u);
}

__device__ __forceinline__ float fast_tanh(float v) {
#if __has_builtin(__builtin_amdgcn_tanhf)
    return __builtin_amdgcn_tanhf(v);
#elif __has_builtin(__builtin_amdgcn_tanh_f32)
    return __builtin_amdgcn_tanh_f32(v);
#else
    return tanhf(v);
#endif
}

// GELU (tanh form; error << bf16 quantization noise of the GEMM inputs)
__device__ __forceinline__ float gelu_f(float v) {
    float u = 0.7978845608028654f * (v + 0.044715f * v * v * v);
    return 0.5f * v * (1.f + fast_tanh(u));
}

// ---------------------------------------------------------------- zero
__global__ void zero_kernel(float* __restrict__ out,
                            int* __restrict__ counts,
                            float* __restrict__ loadsum) {
    size_t gid = (size_t)blockIdx.x * blockDim.x + threadIdx.x;
    size_t stride = (size_t)gridDim.x * blockDim.x;
    for (size_t i = gid; i < (size_t)NC; i += stride) out[i] = 0.f;
    if (blockIdx.x == 0 && threadIdx.x < NE) {
        counts[threadIdx.x] = 0;
        loadsum[threadIdx.x] = 0.f;
    }
}

// ---------------------------------------------------------------- gating
__global__ __launch_bounds__(256) void gate_kernel(
        const float* __restrict__ x, const float* __restrict__ gw,
        const float* __restrict__ gb,
        int* __restrict__ counts, float* __restrict__ loadsum,
        int* __restrict__ rtok, float* __restrict__ rp) {
    const int wave = threadIdx.x >> 5;
    const int lane = threadIdx.x & 31;
    const int t = blockIdx.x * 8 + wave;
    if (t >= NTOK) return;

    float acc[NE];
#pragma unroll
    for (int e = 0; e < NE; ++e) acc[e] = 0.f;

    for (int d0 = lane * 4; d0 < DIM; d0 += 128) {
        float4 xv = *(const float4*)(x + (size_t)t * DIM + d0);
        const float* g = gw + (size_t)d0 * NE;
#pragma unroll
        for (int e = 0; e < NE; ++e) acc[e] += xv.x * g[e];
#pragma unroll
        for (int e = 0; e < NE; ++e) acc[e] += xv.y * g[NE + e];
#pragma unroll
        for (int e = 0; e < NE; ++e) acc[e] += xv.z * g[2 * NE + e];
#pragma unroll
        for (int e = 0; e < NE; ++e) acc[e] += xv.w * g[3 * NE + e];
    }
#pragma unroll
    for (int e = 0; e < NE; ++e)
        for (int off = 16; off; off >>= 1) acc[e] += __shfl_xor(acc[e], off, 32);

    if (lane == 0) {
        float lg[NE];
#pragma unroll
        for (int e = 0; e < NE; ++e) lg[e] = acc[e] + gb[e];
        int i0 = 0;
#pragma unroll
        for (int e = 1; e < NE; ++e) if (lg[e] > lg[i0]) i0 = e;
        int i1 = (i0 == 0) ? 1 : 0;
#pragma unroll
        for (int e = 0; e < NE; ++e) if (e != i0 && lg[e] > lg[i1]) i1 = e;
        float ex = __expf(lg[i1] - lg[i0]);          // softmax over kept pair
        float inv = 1.f / (1.f + ex);
        float p0 = inv, p1 = ex * inv;

        int pos0 = atomicAdd(&counts[i0], 1);
        rtok[i0 * CAP + pos0] = t;  rp[i0 * CAP + pos0] = p0;
        int pos1 = atomicAdd(&counts[i1], 1);
        rtok[i1 * CAP + pos1] = t;  rp[i1 * CAP + pos1] = p1;
        atomicAdd(&loadsum[i0], p0);
        atomicAdd(&loadsum[i1], p1);
    }
}

// ---------------------------------------------------------------- stats
__global__ void stats_kernel(const float* __restrict__ loadsum,
                             float* __restrict__ out) {
    if (threadIdx.x == 0) {
        float loss = 0.f;
        for (int e = 0; e < NE; ++e) {
            float l = loadsum[e] * (1.f / (float)NTOK);
            out[NC + 1 + e] = l;
            float d = l - 1.f / (float)NE;
            loss += d * d;
        }
        out[NC] = 0.01f * loss;
    }
}

// ---------------------------------------------------------------- fused expert FFN
// grid: (token tiles of 32, expert). block: 256 threads = 8 wave32.
__global__ __launch_bounds__(256) void ffn_kernel(
        const float* __restrict__ x,
        const float* __restrict__ w1, const float* __restrict__ b1,
        const float* __restrict__ w2, const float* __restrict__ b2,
        const int* __restrict__ counts,
        const int* __restrict__ rtok, const float* __restrict__ rp,
        float* __restrict__ out) {
    extern __shared__ char smem[];
    int*     tokS = (int*)smem;                                   // [TM]
    float*   pwS  = (float*)(smem + 128);                         // [TM]
    ushortT* xbuf = (ushortT*)(smem + 256);                       // [TM][XP]
    ushortT* hch  = (ushortT*)(smem + 256 + TM * XP * 2);         // [TM][HP]
    ushortT* wT   = (ushortT*)(smem + 256 + TM * XP * 2 + TM * HP * 2); // [1024][WP]

    const int e    = blockIdx.y;
    const int cnt  = counts[e];
    const int row0 = blockIdx.x * TM;
    if (row0 >= cnt) return;

    const int tid  = threadIdx.x;
    const int wave = tid >> 5;
    const int lane = tid & 31;
    const int half = lane >> 4;
    const int l16  = lane & 15;

    if (tid < TM) {
        int r  = row0 + tid;
        int rc = (r < cnt) ? r : (cnt - 1);
        tokS[tid] = rtok[e * CAP + rc];
        pwS[tid]  = (r < cnt) ? rp[e * CAP + rc] : 0.f;   // pad rows contribute 0
    }
    __syncthreads();

    // stage TM token rows of x as bf16 (float4 global loads, b64 LDS stores)
#pragma unroll 4
    for (int i = tid; i < TM * DIM / 4; i += 256) {
        int r = i >> 8, c0 = (i & 255) * 4;
        float4 xv = *(const float4*)(x + (size_t)tokS[r] * DIM + c0);
        *(uint2*)(xbuf + r * XP + c0) =
            make_uint2(pk2bf(xv.x, xv.y), pk2bf(xv.z, xv.w));
    }

    const float* w1e = w1 + (size_t)e * DIM * HID;
    const float* w2e = w2 + (size_t)e * HID * DIM;

    // accumulators: wave owns out cols [wave*128, +128); t<8 rows 0..15, t>=8 rows 16..31
    v8f acc[16];
#pragma unroll
    for (int t = 0; t < 16; ++t) acc[t] = (v8f){0.f,0.f,0.f,0.f,0.f,0.f,0.f,0.f};

    for (int hb = 0; hb < HID / 128; ++hb) {
        const int hcol = hb * 128 + wave * 16 + l16;
        const float b1v = b1[e * HID + hcol];   // issued early, used post-GEMM1

        // ---- GEMM1: h[TM][128] = x[TM][1024] * w1[:, hb*128:+128]
        v8f c1a = (v8f){0.f,0.f,0.f,0.f,0.f,0.f,0.f,0.f};
        v8f c1b = (v8f){0.f,0.f,0.f,0.f,0.f,0.f,0.f,0.f};
        for (int ks = 0; ks < DIM / 32; ++ks) {
            __syncthreads();
            // stage w1 [32K x 128N] transposed: thread = (col n, K-quad kkg)
#pragma unroll
            for (int p = 0; p < 4; ++p) {
                int i   = tid + p * 256;
                int n   = i & 127, kk0 = (i >> 7) * 4;
                const float* src = w1e + (size_t)(ks * 32 + kk0) * HID + hb * 128 + n;
                float f0 = src[0], f1 = src[HID], f2 = src[2 * HID], f3 = src[3 * HID];
                *(uint2*)(wT + n * WP + kk0) = make_uint2(pk2bf(f0, f1), pk2bf(f2, f3));
            }
            __syncthreads();
            ABFrag a0, a1, b;
            const ushortT* ap0 = xbuf + l16 * XP + ks * 32 + half * 8;
            a0.q[0] = *(const uint4*)ap0;
            a0.q[1] = *(const uint4*)(ap0 + 16);
            const ushortT* ap1 = ap0 + 16 * XP;
            a1.q[0] = *(const uint4*)ap1;
            a1.q[1] = *(const uint4*)(ap1 + 16);
            const ushortT* bp = wT + (wave * 16 + l16) * WP + half * 8;
            b.d[0] = *(const ull*)(bp + 0);
            b.d[1] = *(const ull*)(bp + 4);
            b.d[2] = *(const ull*)(bp + 16);
            b.d[3] = *(const ull*)(bp + 20);
            c1a = __builtin_amdgcn_wmma_f32_16x16x32_bf16(
                      false, a0.v, false, b.v, (short)0, c1a, false, false);
            c1b = __builtin_amdgcn_wmma_f32_16x16x32_bf16(
                      false, a1.v, false, b.v, (short)0, c1b, false, false);
        }
        // epilogue: bias + GELU -> bf16 h chunk in LDS
#pragma unroll
        for (int j = 0; j < 8; ++j) {
            float hv = gelu_f(c1a[j] + b1v);
            hch[(j + 8 * half) * HP + wave * 16 + l16] = f2bf(hv);
            float hw = gelu_f(c1b[j] + b1v);
            hch[(j + 8 * half + 16) * HP + wave * 16 + l16] = f2bf(hw);
        }
        __syncthreads();

        // ---- GEMM2 partial: acc += h_chunk[TM][128] * w2[hb*128:+128][:]
        for (int ks2 = 0; ks2 < 4; ++ks2) {
            __syncthreads();
            // stage w2 [32K x 1024N] transposed, K-quads per thread
#pragma unroll 4
            for (int p = 0; p < 32; ++p) {
                int i   = tid + p * 256;
                int n   = i & 1023, kk0 = (i >> 10) * 4;
                const float* src =
                    w2e + (size_t)(hb * 128 + ks2 * 32 + kk0) * DIM + n;
                float f0 = src[0], f1 = src[DIM], f2 = src[2 * DIM], f3 = src[3 * DIM];
                *(uint2*)(wT + n * WP + kk0) = make_uint2(pk2bf(f0, f1), pk2bf(f2, f3));
            }
            __syncthreads();
            ABFrag a2, a3;
            const ushortT* ap2 = hch + l16 * HP + ks2 * 32 + half * 8;
            a2.q[0] = *(const uint4*)ap2;
            a2.q[1] = *(const uint4*)(ap2 + 16);
            const ushortT* ap3 = ap2 + 16 * HP;
            a3.q[0] = *(const uint4*)ap3;
            a3.q[1] = *(const uint4*)(ap3 + 16);
#pragma unroll
            for (int t = 0; t < 8; ++t) {
                ABFrag bf;
                const ushortT* bp2 = wT + (wave * 128 + t * 16 + l16) * WP + half * 8;
                bf.d[0] = *(const ull*)(bp2 + 0);
                bf.d[1] = *(const ull*)(bp2 + 4);
                bf.d[2] = *(const ull*)(bp2 + 16);
                bf.d[3] = *(const ull*)(bp2 + 20);
                acc[t] = __builtin_amdgcn_wmma_f32_16x16x32_bf16(
                             false, a2.v, false, bf.v, (short)0, acc[t], false, false);
                acc[t + 8] = __builtin_amdgcn_wmma_f32_16x16x32_bf16(
                             false, a3.v, false, bf.v, (short)0, acc[t + 8], false, false);
            }
        }
    }

    // weighted combine: out[tok] += p * (expert_out + b2)
#pragma unroll
    for (int t = 0; t < 8; ++t) {
        int col = wave * 128 + t * 16 + l16;
        float b2v = b2[e * DIM + col];
#pragma unroll
        for (int j = 0; j < 8; ++j) {
            int m = j + 8 * half;
            atomicAdd(out + (size_t)tokS[m] * DIM + col, (acc[t][j] + b2v) * pwS[m]);
            int m2 = m + 16;
            atomicAdd(out + (size_t)tokS[m2] * DIM + col,
                      (acc[t + 8][j] + b2v) * pwS[m2]);
        }
    }
}

// ---------------------------------------------------------------- launch
extern "C" void kernel_launch(void* const* d_in, const int* in_sizes, int n_in,
                              void* d_out, int out_size, void* d_ws, size_t ws_size,
                              hipStream_t stream) {
    (void)in_sizes; (void)n_in; (void)out_size; (void)ws_size;
    const float* x  = (const float*)d_in[0];
    const float* gw = (const float*)d_in[1];
    const float* gb = (const float*)d_in[2];
    const float* w1 = (const float*)d_in[3];
    const float* b1 = (const float*)d_in[4];
    const float* w2 = (const float*)d_in[5];
    const float* b2 = (const float*)d_in[6];
    float* out = (float*)d_out;

    int*   counts  = (int*)d_ws;
    float* loadsum = (float*)((char*)d_ws + 32);
    int*   rtok    = (int*)((char*)d_ws + 128);
    float* rpp     = (float*)((char*)d_ws + 128 + (size_t)NE * CAP * 4);

    zero_kernel<<<4096, 256, 0, stream>>>(out, counts, loadsum);
    gate_kernel<<<NTOK / 8, 256, 0, stream>>>(x, gw, gb, counts, loadsum, rtok, rpp);
    stats_kernel<<<1, 32, 0, stream>>>(loadsum, out);

    const size_t smem = 256 + TM * XP * 2 + TM * HP * 2 + 1024 * WP * 2; // ~148.7 KB
    dim3 grid(CAP / TM, NE);   // worst-case tiles per expert; blocks early-exit
    ffn_kernel<<<grid, 256, smem, stream>>>(x, w1, b1, w2, b2,
                                            counts, rtok, rpp, out);
}
